// RoIPooling_3882650436582
// MI455X (gfx1250) — matrix operand order repaired and checked by reading, not compiled, not run on hardware
//
#include <hip/hip_runtime.h>
#include <hip/hip_bf16.h>

// ---- CDNA5 vector types -----------------------------------------------------
typedef __attribute__((ext_vector_type(16))) __bf16        v16bf;
typedef __attribute__((ext_vector_type(8)))  float         v8f;
typedef __attribute__((ext_vector_type(4)))  float         f4;
typedef __attribute__((ext_vector_type(4)))  unsigned int  u4;

#define B_    64
#define C_    2048
#define H_    28
#define W_    28
#define OUT_  4096
#define L_    8
#define K_    16384   // L_ * C_

// fp32 -> bf16 round-to-nearest-even (used in the cold pooling kernel)
__device__ __forceinline__ unsigned short f2bf(float f) {
    unsigned int u = __float_as_uint(f);
    u += 0x7FFFu + ((u >> 16) & 1u);
    return (unsigned short)(u >> 16);
}

// two fp32 -> packed 2 x bf16 (round-half-up): 2x v_add + 1x v_perm_b32.
// D bytes {0,1} = ua bytes {2,3}; D bytes {2,3} = ub bytes {2,3}.
__device__ __forceinline__ unsigned int f2bf_pk(float a, float b) {
    unsigned int ua = __float_as_uint(a) + 0x8000u;
    unsigned int ub = __float_as_uint(b) + 0x8000u;
    return __builtin_amdgcn_perm(ub, ua, 0x07060302u);
}

// ---------------------------------------------------------------------------
// Kernel 1: RoI bilinear sample + 2x2 max pool, emit bf16 X[64, 16384]
// One block per (b, l); threads stride over channels.
// ---------------------------------------------------------------------------
__global__ void __launch_bounds__(256)
roi_pool_bf16(const float* __restrict__ feat,
              const float* __restrict__ lms,
              unsigned short* __restrict__ X) {
    const int b = blockIdx.x;
    const int l = blockIdx.y;

    const float lmx = lms[b * (2 * L_) + 2 * l]     * (7.0f / 224.0f);
    const float lmy = lms[b * (2 * L_) + 2 * l + 1] * (7.0f / 224.0f);
    const float tx  = -1.0f + 2.0f * lmx / 7.0f;
    const float ty  = -1.0f + 2.0f * lmy / 7.0f;
    const float a   = 2.0f / 7.0f;

    int   nidx[4][4];
    float nw[4][4];
    #pragma unroll
    for (int i = 0; i < 2; ++i) {
        const float base_i = (i == 0) ? -0.5f : 0.5f;
        const float gy = a * base_i + ty;
        const float iy = fminf(fmaxf(((gy + 1.0f) * (float)H_ - 1.0f) * 0.5f, 0.0f),
                               (float)(H_ - 1));
        const float y0 = floorf(iy);
        const float wy = iy - y0;
        const int y0i = min(max((int)y0, 0), H_ - 1);
        const int y1i = min(y0i + 1, H_ - 1);
        #pragma unroll
        for (int j = 0; j < 2; ++j) {
            const float base_j = (j == 0) ? -0.5f : 0.5f;
            const float gx = a * base_j + tx;
            const float ix = fminf(fmaxf(((gx + 1.0f) * (float)W_ - 1.0f) * 0.5f, 0.0f),
                                   (float)(W_ - 1));
            const float x0 = floorf(ix);
            const float wx = ix - x0;
            const int x0i = min(max((int)x0, 0), W_ - 1);
            const int x1i = min(x0i + 1, W_ - 1);
            const int p = i * 2 + j;
            nidx[p][0] = y0i * W_ + x0i;  nw[p][0] = (1.0f - wx) * (1.0f - wy);
            nidx[p][1] = y0i * W_ + x1i;  nw[p][1] = wx * (1.0f - wy);
            nidx[p][2] = y1i * W_ + x0i;  nw[p][2] = (1.0f - wx) * wy;
            nidx[p][3] = y1i * W_ + x1i;  nw[p][3] = wx * wy;
        }
    }

    for (int c = threadIdx.x; c < C_; c += blockDim.x) {
        const float* fp = feat + ((size_t)b * C_ + c) * (H_ * W_);
        float m = -3.402823466e38f;
        #pragma unroll
        for (int p = 0; p < 4; ++p) {
            const float v = fp[nidx[p][0]] * nw[p][0] + fp[nidx[p][1]] * nw[p][1]
                          + fp[nidx[p][2]] * nw[p][2] + fp[nidx[p][3]] * nw[p][3];
            m = fmaxf(m, v);
        }
        X[(size_t)b * K_ + l * C_ + c] = f2bf(m);
    }
}

// ---------------------------------------------------------------------------
// Kernel 2: Y[64,4096] = relu(X_bf16 @ W^T + bias) via v_wmma_f32_16x16x32_bf16
// 256 blocks (16 N-cols each) x 128 threads (4 waves). Waves split K (disjoint
// W slices -> W streamed exactly once from HBM, non-temporal). Explicit
// software-pipelined double buffering hides load latency behind cvt+WMMA.
// ---------------------------------------------------------------------------
union AFrag { v16bf v; u4 q[2]; };
union BFrag { v16bf v; unsigned int u[8]; };

struct RawB { f4 f0, f1, f2, f3; };
struct RawA { u4 q0, q1; };

__global__ void __launch_bounds__(128)
gemm_bf16_wmma(const float* __restrict__ Wm,
               const float* __restrict__ bias,
               const unsigned short* __restrict__ X,
               float* __restrict__ Y) {
    __shared__ float red[4 * 4 * 8 * 32];   // [wave][mtile][vgpr][lane] = 16 KB

    const int tid    = threadIdx.x;
    const int lane   = tid & 31;
    const int wave   = tid >> 5;
    const int half   = lane >> 4;       // 0: lanes 0-15, 1: lanes 16-31
    const int nl     = lane & 15;
    const int n_base = blockIdx.x * 16;

    v8f acc0 = {}; v8f acc1 = {}; v8f acc2 = {}; v8f acc3 = {};

    const int k_begin = wave * (K_ / 4);
    const int k_end   = k_begin + (K_ / 4);

    const float*          wrow = Wm + (size_t)(n_base + nl) * K_;
    const unsigned short* xr[4] = {
        X + (size_t)(0 * 16 + nl) * K_,
        X + (size_t)(1 * 16 + nl) * K_,
        X + (size_t)(2 * 16 + nl) * K_,
        X + (size_t)(3 * 16 + nl) * K_
    };

    auto load_raw = [&](int kb, RawB& rb, RawA ra[4]) {
        const f4* wp = (const f4*)(wrow + kb + 16 * half);
        rb.f0 = __builtin_nontemporal_load(wp + 0);
        rb.f1 = __builtin_nontemporal_load(wp + 1);
        rb.f2 = __builtin_nontemporal_load(wp + 2);
        rb.f3 = __builtin_nontemporal_load(wp + 3);
        #pragma unroll
        for (int mt = 0; mt < 4; ++mt) {
            ra[mt].q0 = *(const u4*)(xr[mt] + kb + 16 * half);
            ra[mt].q1 = *(const u4*)(xr[mt] + kb + 16 * half + 8);
        }
    };

    auto compute = [&](const RawB& rb, const RawA ra[4]) {
        BFrag bf;
        bf.u[0] = f2bf_pk(rb.f0.x, rb.f0.y);
        bf.u[1] = f2bf_pk(rb.f0.z, rb.f0.w);
        bf.u[2] = f2bf_pk(rb.f1.x, rb.f1.y);
        bf.u[3] = f2bf_pk(rb.f1.z, rb.f1.w);
        bf.u[4] = f2bf_pk(rb.f2.x, rb.f2.y);
        bf.u[5] = f2bf_pk(rb.f2.z, rb.f2.w);
        bf.u[6] = f2bf_pk(rb.f3.x, rb.f3.y);
        bf.u[7] = f2bf_pk(rb.f3.z, rb.f3.w);

        AFrag a0, a1, a2, a3;
        a0.q[0] = ra[0].q0; a0.q[1] = ra[0].q1;
        a1.q[0] = ra[1].q0; a1.q[1] = ra[1].q1;
        a2.q[0] = ra[2].q0; a2.q[1] = ra[2].q1;
        a3.q[0] = ra[3].q0; a3.q[1] = ra[3].q1;

        acc0 = __builtin_amdgcn_wmma_f32_16x16x32_bf16(false, a0.v, false, bf.v,
                                                       (short)0, acc0, false, false);
        acc1 = __builtin_amdgcn_wmma_f32_16x16x32_bf16(false, a1.v, false, bf.v,
                                                       (short)0, acc1, false, false);
        acc2 = __builtin_amdgcn_wmma_f32_16x16x32_bf16(false, a2.v, false, bf.v,
                                                       (short)0, acc2, false, false);
        acc3 = __builtin_amdgcn_wmma_f32_16x16x32_bf16(false, a3.v, false, bf.v,
                                                       (short)0, acc3, false, false);
    };

    // ---- software pipeline: prologue load, steady loop, peeled epilogue ----
    RawB rb_cur; RawA ra_cur[4];
    load_raw(k_begin, rb_cur, ra_cur);

    for (int kb = k_begin; kb < k_end - 32; kb += 32) {
        RawB rb_nxt; RawA ra_nxt[4];
        load_raw(kb + 32, rb_nxt, ra_nxt);     // in flight during compute below
        compute(rb_cur, ra_cur);
        rb_cur = rb_nxt;
        #pragma unroll
        for (int mt = 0; mt < 4; ++mt) ra_cur[mt] = ra_nxt[mt];
    }
    compute(rb_cur, ra_cur);                   // peeled last step (no OOB load)

    // ---- cross-wave K reduction through LDS --------------------------------
    #pragma unroll
    for (int r = 0; r < 8; ++r) {
        red[((wave * 4 + 0) * 8 + r) * 32 + lane] = acc0[r];
        red[((wave * 4 + 1) * 8 + r) * 32 + lane] = acc1[r];
        red[((wave * 4 + 2) * 8 + r) * 32 + lane] = acc2[r];
        red[((wave * 4 + 3) * 8 + r) * 32 + lane] = acc3[r];
    }
    __syncthreads();

    // 1024 outputs, 128 threads -> 8 each. D layout: lanes 0-15 M=r, 16-31 M=8+r
    #pragma unroll
    for (int i = 0; i < 8; ++i) {
        const int flat = tid + i * 128;     // (mt, r, lane)
        const int lan  = flat & 31;
        const int r    = (flat >> 5) & 7;
        const int mt   = flat >> 8;
        float s = red[((0 * 4 + mt) * 8 + r) * 32 + lan]
                + red[((1 * 4 + mt) * 8 + r) * 32 + lan]
                + red[((2 * 4 + mt) * 8 + r) * 32 + lan]
                + red[((3 * 4 + mt) * 8 + r) * 32 + lan];
        const int m = mt * 16 + ((lan < 16) ? r : r + 8);
        const int n = n_base + (lan & 15);
        s += bias[n];
        Y[(size_t)m * OUT_ + n] = fmaxf(s, 0.0f);
    }
}

// ---------------------------------------------------------------------------
extern "C" void kernel_launch(void* const* d_in, const int* in_sizes, int n_in,
                              void* d_out, int out_size, void* d_ws, size_t ws_size,
                              hipStream_t stream) {
    const float* feat = (const float*)d_in[0];   // [64,2048,28,28]
    const float* lms  = (const float*)d_in[1];   // [64,16]
    const float* Wm   = (const float*)d_in[2];   // [4096,16384]
    const float* bias = (const float*)d_in[3];   // [4096]

    unsigned short* Xbf = (unsigned short*)d_ws; // bf16 X[64,16384] = 2 MB

    roi_pool_bf16<<<dim3(B_, L_), 256, 0, stream>>>(feat, lms, Xbf);
    gemm_bf16_wmma<<<dim3(OUT_ / 16), 128, 0, stream>>>(Wm, bias, Xbf, (float*)d_out);
}
